// SSMCore_32899449487950
// MI455X (gfx1250) — matrix operand order repaired and checked by reading, not compiled, not run on hardware
//
#include <hip/hip_runtime.h>

// ---------------------------------------------------------------------------
// SSM core for MI455X (gfx1250): spectral-norm + WMMA bf16 GEMMs + serial scan
// ---------------------------------------------------------------------------

#define BATCH 32
#define SEQN  4096
#define CIN   128
#define HID   256
#define COUT  128
#define ROWS  (BATCH * SEQN)   // 131072

typedef __attribute__((ext_vector_type(16))) __bf16 v16bf;
typedef __attribute__((ext_vector_type(16))) short  v16s;
typedef __attribute__((ext_vector_type(8)))  float  v8f;
typedef __attribute__((ext_vector_type(4)))  int    v4i;

// ---- f32 -> bf16 conversion helpers (prefer native HW convert) -------------
__device__ __forceinline__ unsigned short f2bf_bits(float f) {
    // fptrunc -> native v_cvt_*bf16* when the target has it (gfx1250 does)
    __bf16 b = (__bf16)f;
    return __builtin_bit_cast(unsigned short, b);
}
__device__ __forceinline__ unsigned f2bf_pack(float a, float b) {
#if defined(__has_builtin) && __has_builtin(__builtin_amdgcn_cvt_pk_bf16_f32)
    auto p = __builtin_amdgcn_cvt_pk_bf16_f32(a, b);
    return __builtin_bit_cast(unsigned, p);
#else
    typedef __attribute__((ext_vector_type(2))) __bf16 v2bf;
    v2bf p;
    p[0] = (__bf16)a;   // clang vectorizes the pair into v_cvt_pk_bf16_f32
    p[1] = (__bf16)b;   // when available
    return __builtin_bit_cast(unsigned, p);
#endif
}
__device__ __forceinline__ float bf2f(unsigned short s) {
    return __builtin_bit_cast(float, (unsigned)s << 16);
}

// ---- async global->LDS copy (CDNA5), guarded with sync fallback ------------
#if defined(__has_builtin)
#if __has_builtin(__builtin_amdgcn_global_load_async_to_lds_b128)
#define USE_ASYNC_LDS 1
#endif
#endif

#ifdef USE_ASYNC_LDS
__device__ __forceinline__ void g2l_b128(void* l, const void* g) {
    auto gp = (__attribute__((address_space(1))) v4i*)(const_cast<void*>(g));
    auto lp = (__attribute__((address_space(3))) v4i*)(l);
    __builtin_amdgcn_global_load_async_to_lds_b128(gp, lp, /*offset=*/0, /*cpol=*/0);
}
__device__ __forceinline__ void async_wait0() {
#if __has_builtin(__builtin_amdgcn_s_wait_asynccnt)
    __builtin_amdgcn_s_wait_asynccnt(0);
#else
    asm volatile("s_wait_asynccnt 0x0" ::: "memory");
#endif
}
#endif

// A-fragment (16x32 bf16, MxK): lane holds row M=lane&15; K pairs contiguous,
// hi half-wave gets K+16 (matches load_bfrag).
__device__ __forceinline__ v16bf load_afrag(const unsigned short* row, int hi) {
    v16s a;
#pragma unroll
    for (int p = 0; p < 8; ++p) {
        int k = (hi << 4) + (p << 1);
        unsigned w = *(const unsigned*)(row + k);
        a[2 * p]     = (short)(w & 0xffffu);
        a[2 * p + 1] = (short)(w >> 16);
    }
    return __builtin_bit_cast(v16bf, a);
}

// B-fragment (32x16 bf16, KxN) built from B^T stored [N][K] row-major:
// lane holds column N=lane&15; K pairs contiguous, hi half-wave gets K+16.
__device__ __forceinline__ v16bf load_bfrag(const unsigned short* rowN, int hi) {
    v16s b;
#pragma unroll
    for (int v = 0; v < 8; ++v) {
        int k = (hi << 4) + (v << 1);
        unsigned w = *(const unsigned*)(rowN + k);
        b[2 * v]     = (short)(w & 0xffffu);
        b[2 * v + 1] = (short)(w >> 16);
    }
    return __builtin_bit_cast(v16bf, b);
}

__device__ __forceinline__ v8f wmma_bf16(v16bf a, v16bf b, v8f c) {
    return __builtin_amdgcn_wmma_f32_16x16x32_bf16(
        /*neg_a=*/false, a, /*neg_b=*/false, b,
        /*c_mod=*/(short)0, c, /*reuse_a=*/false, /*reuse_b=*/false);
}

// ---------------------------------------------------------------------------
// Kernel 1: spectral norm of A (256x256) via power iteration on A^T A.
// Writes scale = 1/(sigma + 1e-5).
// ---------------------------------------------------------------------------
__global__ __launch_bounds__(256) void signorm_kernel(const float* __restrict__ A,
                                                      float* __restrict__ scale) {
    __shared__ float v[HID];
    __shared__ float w[HID];
    __shared__ float red[HID];
    const int i = threadIdx.x;
    v[i] = 1.0f;
    __syncthreads();
    for (int it = 0; it < 30; ++it) {
        float s = 0.f;
        for (int c = 0; c < HID; ++c) s += A[i * HID + c] * v[c];
        w[i] = s;
        __syncthreads();
        float s2 = 0.f;
        for (int r = 0; r < HID; ++r) s2 += A[r * HID + i] * w[r];
        red[i] = s2 * s2;
        __syncthreads();
        for (int off = 128; off > 0; off >>= 1) {
            if (i < off) red[i] += red[i + off];
            __syncthreads();
        }
        float inv = rsqrtf(red[0] + 1e-30f);
        __syncthreads();
        v[i] = s2 * inv;
        __syncthreads();
    }
    float s = 0.f;
    for (int c = 0; c < HID; ++c) s += A[i * HID + c] * v[c];
    red[i] = s * s;
    __syncthreads();
    for (int off = 128; off > 0; off >>= 1) {
        if (i < off) red[i] += red[i + off];
        __syncthreads();
    }
    if (i == 0) scale[0] = 1.0f / (sqrtf(red[0]) + 1e-5f);
}

// ---------------------------------------------------------------------------
// Kernel 2: bake scale into bf16 A_hat (row-major = B^T layout for h@A^T),
// convert W_B [HID][CIN] and W_C [COUT][HID] to bf16 (already [N][K] major).
// ---------------------------------------------------------------------------
__global__ __launch_bounds__(256) void prep_kernel(const float* __restrict__ A,
                                                   const float* __restrict__ Wb,
                                                   const float* __restrict__ Wc,
                                                   const float* __restrict__ scale,
                                                   unsigned short* __restrict__ Ahat,
                                                   unsigned short* __restrict__ Wbb,
                                                   unsigned short* __restrict__ Wcb) {
    const int idx = blockIdx.x * blockDim.x + threadIdx.x;
    const float sc = scale[0];
    if (idx < HID * HID) Ahat[idx] = f2bf_bits(A[idx] * sc);
    if (idx < HID * CIN) Wbb[idx] = f2bf_bits(Wb[idx]);
    if (idx < COUT * HID) Wcb[idx] = f2bf_bits(Wc[idx]);
}

// ---------------------------------------------------------------------------
// Kernel 3: u = x @ W_B^T   (131072x128) x (128x256) -> bf16 u [b*t][HID]
// 256 threads / 8 waves; workgroup tile 128(M) x 64(N); K chunk 32 via LDS.
// ---------------------------------------------------------------------------
#define LPAD 8
#define LSTR (32 + LPAD)   // 40 u16 = 80B row stride (16B aligned, conflict-spread)

__global__ __launch_bounds__(256) void uproj_kernel(const float* __restrict__ x,
                                                    const unsigned short* __restrict__ Wb,
                                                    unsigned short* __restrict__ u) {
    __shared__ unsigned short As[128 * LSTR];
    __shared__ unsigned short Ws[64 * LSTR];
    const int tid = threadIdx.x, lane = tid & 31, wave = tid >> 5;
    const int hi = lane >> 4, nl = lane & 15;
    const long r0 = (long)blockIdx.x * 128;
    const int n0 = blockIdx.y * 64;

    v8f acc[4];
#pragma unroll
    for (int j = 0; j < 4; ++j)
#pragma unroll
        for (int i = 0; i < 8; ++i) acc[j][i] = 0.f;

    for (int k0 = 0; k0 < CIN; k0 += 32) {
        __syncthreads();
        // stage W tile 64x32 bf16 (pure copy -> async to LDS on CDNA5)
        {
            int row = tid >> 2, seg = tid & 3;
            const unsigned short* src = Wb + (n0 + row) * CIN + k0 + seg * 8;
            unsigned short* dst = Ws + row * LSTR + seg * 8;
#ifdef USE_ASYNC_LDS
            g2l_b128(dst, src);
#else
            *(uint4*)dst = *(const uint4*)src;
#endif
        }
        // stage A tile 128x32: f32 -> bf16 (conversion en route, stays VALU)
#pragma unroll
        for (int pass = 0; pass < 4; ++pass) {
            int row = pass * 32 + (tid >> 3);
            int seg = tid & 7;
            float4 xv = *(const float4*)(x + (r0 + row) * CIN + k0 + seg * 4);
            unsigned short* dst = As + row * LSTR + seg * 4;
            *(unsigned*)(dst)     = f2bf_pack(xv.x, xv.y);
            *(unsigned*)(dst + 2) = f2bf_pack(xv.z, xv.w);
        }
#ifdef USE_ASYNC_LDS
        async_wait0();
#endif
        __syncthreads();
        v16bf af = load_afrag(As + (wave * 16 + nl) * LSTR, hi);
#pragma unroll
        for (int j = 0; j < 4; ++j) {
            v16bf bf = load_bfrag(Ws + (j * 16 + nl) * LSTR, hi);
            acc[j] = wmma_bf16(af, bf, acc[j]);
        }
    }
#pragma unroll
    for (int j = 0; j < 4; ++j)
#pragma unroll
        for (int i = 0; i < 8; i += 2) {
            unsigned pk = f2bf_pack(acc[j][i], acc[j][i + 1]);
            int m = wave * 16 + i + hi * 8;
            int n = n0 + j * 16 + nl;
            u[(r0 + m) * HID + n]     = (unsigned short)(pk & 0xffffu);
            u[(r0 + m + 1) * HID + n] = (unsigned short)(pk >> 16);
        }
}

// ---------------------------------------------------------------------------
// Kernel 4: the serial scan. 2 workgroups x 512 threads (16 waves).
// Each wave owns one 16(batch)x16(hidden) tile of h; A_norm^T B-fragments are
// preloaded into 64 VGPRs once; inner loop = two independent 4-deep WMMA
// chains per step (acc0 C-init = u_t, acc1 C-init = 0, merged with v_add_f32).
// ---------------------------------------------------------------------------
#define HSTR 264   // 16 rows x (256+8) u16 — padded h row

__global__ __launch_bounds__(512) void scan_kernel(const unsigned short* __restrict__ u,
                                                   const unsigned short* __restrict__ Ahat,
                                                   unsigned short* __restrict__ hs) {
    __shared__ unsigned short Hs[2][16 * HSTR];
    const int tid = threadIdx.x, lane = tid & 31, wave = tid >> 5;
    const int hi = lane >> 4, nl = lane & 15;
    const int B0 = blockIdx.x * 16;
    const int n0 = wave * 16;

    // Preload the 8 K-chunk B-fragments of A_norm^T for this wave's N-tile.
    v16bf bfr[8];
#pragma unroll
    for (int c = 0; c < 8; ++c)
        bfr[c] = load_bfrag(Ahat + (n0 + nl) * HID + c * 32, hi);

    // zero h0
    for (int i = tid; i < 16 * HSTR; i += 512) Hs[0][i] = 0;

    // per-lane element offsets for the 8 C/D rows (m = i + hi*8)
    long ubase[8];
#pragma unroll
    for (int i = 0; i < 8; ++i) {
        int m = i + hi * 8;
        ubase[i] = ((long)(B0 + m) * SEQN) * HID + n0 + nl;
    }
    __syncthreads();

    float ucur[8];
#pragma unroll
    for (int i = 0; i < 8; ++i) ucur[i] = bf2f(u[ubase[i]]);

    int buf = 0;
    for (int t = 0; t < SEQN; ++t) {
        // prefetch u a few steps ahead (global_prefetch_b8), pipeline t+1 loads
        int tp = t + 8 < SEQN ? t + 8 : SEQN - 1;
        __builtin_prefetch(u + ubase[0] + (long)tp * HID, 0, 1);
        int tn = t + 1 < SEQN ? t + 1 : t;
        float unext[8];
#pragma unroll
        for (int i = 0; i < 8; ++i) unext[i] = bf2f(u[ubase[i] + (long)tn * HID]);

        v8f acc0, acc1;
#pragma unroll
        for (int i = 0; i < 8; ++i) { acc0[i] = ucur[i]; acc1[i] = 0.f; }

        const unsigned short* hrow = Hs[buf] + nl * HSTR;
#pragma unroll
        for (int c = 0; c < 4; ++c) {
            v16bf af0 = load_afrag(hrow + c * 32, hi);
            v16bf af1 = load_afrag(hrow + (c + 4) * 32, hi);
            acc0 = wmma_bf16(af0, bfr[c], acc0);
            acc1 = wmma_bf16(af1, bfr[c + 4], acc1);
        }

        // merge chains, relu, write h_t to other LDS buffer + stream to global
        unsigned short hb[8];
#pragma unroll
        for (int i = 0; i < 8; i += 2) {
            float a = fmaxf(acc0[i] + acc1[i], 0.f);
            float b = fmaxf(acc0[i + 1] + acc1[i + 1], 0.f);
            unsigned pk = f2bf_pack(a, b);
            hb[i]     = (unsigned short)(pk & 0xffffu);
            hb[i + 1] = (unsigned short)(pk >> 16);
        }
#pragma unroll
        for (int i = 0; i < 8; ++i) {
            int m = i + hi * 8;
            Hs[buf ^ 1][m * HSTR + n0 + nl] = hb[i];
            hs[ubase[i] + (long)t * HID] = hb[i];
        }
#pragma unroll
        for (int i = 0; i < 8; ++i) ucur[i] = unext[i];
        buf ^= 1;
        __syncthreads();
    }
}

// ---------------------------------------------------------------------------
// Kernel 5: y = hs @ W_C^T  (131072x256) x (256x128) -> f32 d_out
// ---------------------------------------------------------------------------
__global__ __launch_bounds__(256) void yproj_kernel(const unsigned short* __restrict__ hsb,
                                                    const unsigned short* __restrict__ Wc,
                                                    float* __restrict__ y) {
    __shared__ unsigned short As[128 * LSTR];
    __shared__ unsigned short Ws[64 * LSTR];
    const int tid = threadIdx.x, lane = tid & 31, wave = tid >> 5;
    const int hi = lane >> 4, nl = lane & 15;
    const long r0 = (long)blockIdx.x * 128;
    const int n0 = blockIdx.y * 64;

    v8f acc[4];
#pragma unroll
    for (int j = 0; j < 4; ++j)
#pragma unroll
        for (int i = 0; i < 8; ++i) acc[j][i] = 0.f;

    for (int k0 = 0; k0 < HID; k0 += 32) {
        __syncthreads();
        // stage A tile 128x32 bf16 (2 threads per row, two b128 each)
        {
            int row = tid >> 1, half = tid & 1;
            const unsigned short* src = hsb + (r0 + row) * HID + k0 + half * 16;
            unsigned short* dst = As + row * LSTR + half * 16;
#ifdef USE_ASYNC_LDS
            g2l_b128(dst, src);
            g2l_b128(dst + 8, src + 8);
#else
            *(uint4*)(dst)     = *(const uint4*)(src);
            *(uint4*)(dst + 8) = *(const uint4*)(src + 8);
#endif
        }
        // stage W tile 64x32 bf16
        {
            int row = tid >> 2, seg = tid & 3;
            const unsigned short* src = Wc + (n0 + row) * HID + k0 + seg * 8;
            unsigned short* dst = Ws + row * LSTR + seg * 8;
#ifdef USE_ASYNC_LDS
            g2l_b128(dst, src);
#else
            *(uint4*)dst = *(const uint4*)src;
#endif
        }
#ifdef USE_ASYNC_LDS
        async_wait0();
#endif
        __syncthreads();
        v16bf af = load_afrag(As + (wave * 16 + nl) * LSTR, hi);
#pragma unroll
        for (int j = 0; j < 4; ++j) {
            v16bf bf = load_bfrag(Ws + (j * 16 + nl) * LSTR, hi);
            acc[j] = wmma_bf16(af, bf, acc[j]);
        }
    }
#pragma unroll
    for (int j = 0; j < 4; ++j)
#pragma unroll
        for (int i = 0; i < 8; ++i) {
            int m = wave * 16 + i + hi * 8;
            int n = n0 + j * 16 + nl;
            y[(r0 + m) * COUT + n] = acc[j][i];
        }
}

// ---------------------------------------------------------------------------
extern "C" void kernel_launch(void* const* d_in, const int* in_sizes, int n_in,
                              void* d_out, int out_size, void* d_ws, size_t ws_size,
                              hipStream_t stream) {
    (void)in_sizes; (void)n_in; (void)out_size; (void)ws_size;
    const float* x  = (const float*)d_in[0];
    const float* A  = (const float*)d_in[1];
    const float* Wb = (const float*)d_in[2];
    const float* Wc = (const float*)d_in[3];
    float* y = (float*)d_out;

    char* ws = (char*)d_ws;
    float* scale            = (float*)ws;                              // 256 B
    unsigned short* Ahat    = (unsigned short*)(ws + 256);             // 128 KB
    unsigned short* Wbb     = Ahat + (size_t)HID * HID;                // 64 KB
    unsigned short* Wcb     = Wbb + (size_t)HID * CIN;                 // 64 KB
    unsigned short* u       = Wcb + (size_t)COUT * HID;                // 64 MB
    unsigned short* hsb     = u + (size_t)ROWS * HID;                  // 64 MB

    signorm_kernel<<<dim3(1), dim3(256), 0, stream>>>(A, scale);
    prep_kernel<<<dim3(HID * HID / 256), dim3(256), 0, stream>>>(A, Wb, Wc, scale,
                                                                 Ahat, Wbb, Wcb);
    uproj_kernel<<<dim3(ROWS / 128, HID / 64), dim3(256), 0, stream>>>(x, Wbb, u);
    scan_kernel<<<dim3(BATCH / 16), dim3(512), 0, stream>>>(u, Ahat, hsb);
    yproj_kernel<<<dim3(ROWS / 128, COUT / 64), dim3(256), 0, stream>>>(hsb, Wcb, y);
}
